// DMSADLoss_43860206027138
// MI455X (gfx1250) — compile-verified, matrix operand order checked by reading, not run on hardware
//
#include <hip/hip_runtime.h>

// ---------------------------------------------------------------------------
// DMSAD loss: mean over B of  f(min_j ||x_i - c_j||^2)
// d2 = x2 - 2*(x@c^T) + c2 ; cross term via V_WMMA_F32_16X16X32_BF16,
// x2/c2 kept in exact fp32.  B=131072, C=128, D=256 (C,D hardcoded).
// ---------------------------------------------------------------------------

typedef __attribute__((ext_vector_type(16))) __bf16 v16bf;
typedef __attribute__((ext_vector_type(4)))  __bf16 v4bf;
typedef __attribute__((ext_vector_type(8)))  float  v8f;
typedef __attribute__((ext_vector_type(4)))  float  v4f;

constexpr int D_DIM = 256;   // feature dim
constexpr int C_DIM = 128;   // clusters
constexpr int BROWS = 128;   // rows per workgroup (8 waves x 16 rows)
constexpr int KCH   = 64;    // K chunk staged in LDS
constexpr int LSTR  = 80;    // LDS row stride in bf16 elems (64 + 16 pad -> 160B)
constexpr int NT    = 8;     // N tiles = C_DIM / 16
constexpr float ETA_F = 1.0f;
constexpr float EPS_F = 1e-6f;

__global__ __launch_bounds__(256)
void dmsad_main(const float* __restrict__ X, const float* __restrict__ Cm,
                const long long* __restrict__ st, float* __restrict__ part,
                int B)
{
  __shared__ v4bf  Xs[BROWS * LSTR / 4];   // 20 KB, 160B row stride
  __shared__ v4bf  Cs[C_DIM * LSTR / 4];   // 20 KB
  __shared__ float x2s[BROWS];
  __shared__ float c2s[C_DIM];
  __shared__ float waveSums[8];

  const int tid      = threadIdx.x;
  const int lane     = tid & 31;
  const int wave     = tid >> 5;
  const int h        = lane >> 4;      // lane half (0/1)
  const int lrow     = lane & 15;
  const int rowBase  = blockIdx.x * BROWS;
  const int waveRow0 = wave * 16;

  // Loader mapping: thread owns row (tid>>1), columns [(tid&1)*32 .. +31].
  const int ldRow = tid >> 1;
  const int ldCol = (tid & 1) * 32;
  const int gXrow = rowBase + ldRow;
  const float* xsrc = X  + (size_t)gXrow * D_DIM + ldCol;
  const float* csrc = Cm + (size_t)ldRow * D_DIM + ldCol;
  v4bf* xdst = &Xs[(ldRow * LSTR + ldCol) / 4];
  v4bf* cdst = &Cs[(ldRow * LSTR + ldCol) / 4];

  v8f acc[NT];
  #pragma unroll
  for (int t = 0; t < NT; ++t) acc[t] = (v8f){0.f,0.f,0.f,0.f,0.f,0.f,0.f,0.f};

  float x2p = 0.0f;   // fp32 partial of ||x_row||^2 over this thread's columns
  float c2p = 0.0f;   // fp32 partial of ||c_row||^2

  const int NCH = D_DIM / KCH;
  for (int kc = 0; kc < NCH; ++kc) {
    const int kb = kc * KCH;

    // ---- stage X chunk (fp32 -> bf16 via hw cvt) + exact x2 partial ----
    if (gXrow < B) {
      #pragma unroll
      for (int i = 0; i < 8; ++i) {
        v4f v = reinterpret_cast<const v4f*>(xsrc + kb)[i];
        x2p = __builtin_fmaf(v.x, v.x, x2p);
        x2p = __builtin_fmaf(v.y, v.y, x2p);
        x2p = __builtin_fmaf(v.z, v.z, x2p);
        x2p = __builtin_fmaf(v.w, v.w, x2p);
        xdst[i] = __builtin_convertvector(v, v4bf);   // v_cvt_pk_bf16_f32 x2
      }
      if (kc + 1 < NCH) __builtin_prefetch(xsrc + kb + KCH, 0, 1);
    } else {
      #pragma unroll
      for (int i = 0; i < 8; ++i)
        xdst[i] = (v4bf){(__bf16)0.f, (__bf16)0.f, (__bf16)0.f, (__bf16)0.f};
    }
    // ---- stage C chunk (fp32 -> bf16) + exact c2 partial ----
    #pragma unroll
    for (int i = 0; i < 8; ++i) {
      v4f v = reinterpret_cast<const v4f*>(csrc + kb)[i];
      c2p = __builtin_fmaf(v.x, v.x, c2p);
      c2p = __builtin_fmaf(v.y, v.y, c2p);
      c2p = __builtin_fmaf(v.z, v.z, c2p);
      c2p = __builtin_fmaf(v.w, v.w, c2p);
      cdst[i] = __builtin_convertvector(v, v4bf);
    }
    __syncthreads();

    // ---- WMMA: acc[t] += Xtile(16xK) @ Ctile^T(Kx16) ----
    // A lane (row = lrow, half h) reads K[h*16 .. h*16+15] contiguously;
    // B lane (col = lrow, half h) reads the same K slots from c's row.
    // Identical per-lane K packing on A and B makes any internal K
    // permutation cancel in the dot product.
    const v4bf* aBase = &Xs[(waveRow0 + lrow) * LSTR / 4];
    #pragma unroll
    for (int ks = 0; ks < KCH / 32; ++ks) {
      v16bf a = *reinterpret_cast<const v16bf*>(aBase + ks * 8 + h * 4);
      #pragma unroll
      for (int t = 0; t < NT; ++t) {
        v16bf b = *reinterpret_cast<const v16bf*>(
            &Cs[(t * 16 + lrow) * LSTR / 4 + ks * 8 + h * 4]);
        acc[t] = __builtin_amdgcn_wmma_f32_16x16x32_bf16(
            false, a, false, b, (short)0, acc[t], false, false);
      }
    }
    __syncthreads();
  }

  // ---- deterministic x2/c2 finalize (pair via xor-shuffle, even lane writes)
  float x2o = __shfl_xor(x2p, 1, 32);
  float c2o = __shfl_xor(c2p, 1, 32);
  if ((tid & 1) == 0) {
    x2s[ldRow] = x2p + x2o;
    c2s[ldRow] = c2p + c2o;
  }
  __syncthreads();

  // ---- per-lane min over its 8 columns for its 8 rows ----
  // C/D layout: lane holds N = lrow (+16t per tile), rows r + 8*h (r = vgpr idx)
  float pr[8];
  #pragma unroll
  for (int r = 0; r < 8; ++r) pr[r] = 3.4e38f;
  #pragma unroll
  for (int t = 0; t < NT; ++t) {
    float c2v = c2s[t * 16 + lrow];
    #pragma unroll
    for (int r = 0; r < 8; ++r)
      pr[r] = fminf(pr[r], __builtin_fmaf(-2.0f, acc[t][r], c2v));
  }
  // min across the 16 lanes of each half (covers all 128 columns)
  #pragma unroll
  for (int m = 1; m <= 8; m <<= 1) {
    #pragma unroll
    for (int r = 0; r < 8; ++r)
      pr[r] = fminf(pr[r], __shfl_xor(pr[r], m, 32));
  }

  // lane with lrow == r owns row m = r + 8*h
  float myMin = pr[0];
  #pragma unroll
  for (int r = 1; r < 8; ++r) myMin = (lrow == r) ? pr[r] : myMin;

  float contrib = 0.0f;
  if (lrow < 8) {
    const int m    = lrow + (h << 3);
    const int grow = rowBase + waveRow0 + m;
    if (grow < B) {
      float dist = fmaxf(x2s[waveRow0 + m] + myMin, 0.0f);
      long long s = st[grow];
      float loss;
      if (s == 0)      loss = dist;                    // power not applied
      else if (s > 0)  loss = ETA_F * (dist + EPS_F);  // (d+eps)^{+1}
      else             loss = ETA_F / (dist + EPS_F);  // (d+eps)^{-1}
      contrib = loss;
    }
  }
  // wave sum (deterministic xor tree)
  #pragma unroll
  for (int m = 16; m >= 1; m >>= 1) contrib += __shfl_xor(contrib, m, 32);
  if (lane == 0) waveSums[wave] = contrib;
  __syncthreads();
  if (tid == 0) {
    float s = 0.0f;
    #pragma unroll
    for (int w = 0; w < 8; ++w) s += waveSums[w];
    part[blockIdx.x] = s;
  }
}

__global__ __launch_bounds__(256)
void dmsad_reduce(const float* __restrict__ part, float* __restrict__ out,
                  int n, float invB)
{
  __shared__ float s[256];
  float v = 0.0f;
  for (int i = threadIdx.x; i < n; i += 256) v += part[i];
  s[threadIdx.x] = v;
  __syncthreads();
  for (int off = 128; off > 0; off >>= 1) {
    if (threadIdx.x < off) s[threadIdx.x] += s[threadIdx.x + off];
    __syncthreads();
  }
  if (threadIdx.x == 0) out[0] = s[0] * invB;
}

extern "C" void kernel_launch(void* const* d_in, const int* in_sizes, int n_in,
                              void* d_out, int out_size, void* d_ws, size_t ws_size,
                              hipStream_t stream) {
  const float*     X  = (const float*)d_in[0];       // [B, 256] fp32
  const float*     Cm = (const float*)d_in[1];       // [128, 256] fp32
  const long long* st = (const long long*)d_in[2];   // [B] int64
  float* out = (float*)d_out;
  float* ws  = (float*)d_ws;

  const int B = in_sizes[0] / D_DIM;
  const int nblocks = (B + BROWS - 1) / BROWS;       // 1024 for B=131072

  dmsad_main<<<nblocks, 256, 0, stream>>>(X, Cm, st, ws, B);
  dmsad_reduce<<<1, 256, 0, stream>>>(ws, out, nblocks, 1.0f / (float)B);
}
